// SMDAdapter_29308856827962
// MI455X (gfx1250) — matrix operand order repaired and checked by reading, not compiled, not run on hardware
//
#include <hip/hip_runtime.h>
#include <hip/hip_bf16.h>
#include <math.h>
#include <stdint.h>

typedef __bf16 bf16_t;
typedef __attribute__((ext_vector_type(16))) __bf16 v16bf;
typedef __attribute__((ext_vector_type(8)))  float  v8f;

#define NEGV (-1e9f)
static constexpr int N_TOK = 2048;
static constexpr int BN    = 8192;   // B*N
static constexpr int D_HID = 1024;

#if __has_builtin(__builtin_amdgcn_tensor_load_to_lds) && __has_builtin(__builtin_amdgcn_s_wait_tensorcnt)
#define HAVE_TDM 1
typedef unsigned int v4u __attribute__((ext_vector_type(4)));
typedef int          v4i __attribute__((ext_vector_type(4)));
typedef int          v8i __attribute__((ext_vector_type(8)));
#endif

// ---------------------------------------------------------------- WMMA helpers
__device__ __forceinline__ v8f wmma_bf16(v16bf a, v16bf b, v8f c) {
  return __builtin_amdgcn_wmma_f32_16x16x32_bf16(false, a, false, b, (short)0, c, false, false);
}

// A-matrix (16x32 bf16) element K index for lane-half h, vgpr g, sub j
__device__ __forceinline__ int a_kidx(int g, int j, int h) {
  return ((g & 4) << 2) + 8 * h + 2 * (g & 3) + j;   // (g<4?0:16)+8h+2(g&3)+j
}
// B-matrix (32x16 bf16) element K index
__device__ __forceinline__ int b_kidx(int g, int j, int h) {
  return ((g & 4) << 2) + 8 * h + 2 * (g & 3) + j;
}

// A fragment from row-major bf16 (row stride ld), rows 0..15, K 0..31 at base
__device__ __forceinline__ v16bf load_a_bf16(const bf16_t* base, int ld) {
  const int lane = threadIdx.x & 31, r = lane & 15, h = lane >> 4;
  v16bf a;
#pragma unroll
  for (int g = 0; g < 8; ++g)
#pragma unroll
    for (int j = 0; j < 2; ++j)
      a[2 * g + j] = base[r * ld + a_kidx(g, j, h)];
  return a;
}

// A fragment from row-major f32 (LDS), converting to bf16
__device__ __forceinline__ v16bf load_a_f32cvt(const float* base, int ld) {
  const int lane = threadIdx.x & 31, r = lane & 15, h = lane >> 4;
  v16bf a;
#pragma unroll
  for (int g = 0; g < 8; ++g)
#pragma unroll
    for (int j = 0; j < 2; ++j)
      a[2 * g + j] = (bf16_t)base[r * ld + a_kidx(g, j, h)];
  return a;
}

// B fragment from bf16 matrix used transposed: b(k,n) = M[(row0+n)*ld + k0+k]
__device__ __forceinline__ v16bf load_bT_bf16(const bf16_t* __restrict__ M, int ld,
                                              int row0, int k0) {
  const int lane = threadIdx.x & 31, n = lane & 15, h = lane >> 4;
  v16bf b;
#pragma unroll
  for (int g = 0; g < 8; ++g)
#pragma unroll
    for (int j = 0; j < 2; ++j) {
      const int k = k0 + b_kidx(g, j, h);
      b[2 * g + j] = M[(row0 + n) * ld + k];
    }
  return b;
}

// B fragment non-transposed bf16: b(k,n) = M[(k0+k)*ld + col0+n]
__device__ __forceinline__ v16bf load_b_bf16(const bf16_t* __restrict__ M, int ld,
                                             int k0, int col0) {
  const int lane = threadIdx.x & 31, n = lane & 15, h = lane >> 4;
  v16bf b;
#pragma unroll
  for (int g = 0; g < 8; ++g)
#pragma unroll
    for (int j = 0; j < 2; ++j) {
      const int k = b_kidx(g, j, h);
      b[2 * g + j] = M[(k0 + k) * ld + col0 + n];
    }
  return b;
}

// compat tile [16x16] = Q[16,64] x K[16,64]^T with pre-loaded Q fragments
__device__ __forceinline__ v8f qk_tile_pre(v16bf qa0, v16bf qa1,
                                           const bf16_t* __restrict__ Kb, int rk) {
  v16bf b0 = load_bT_bf16(Kb, 64, rk, 0);
  v16bf b1 = load_bT_bf16(Kb, 64, rk, 32);
  v8f c = {0.f, 0.f, 0.f, 0.f, 0.f, 0.f, 0.f, 0.f};
  c = wmma_bf16(qa0, b0, c);
  c = wmma_bf16(qa1, b1, c);
  return c;
}

// 16-lane (half-wave) reductions; offsets <16 never cross wave halves
__device__ __forceinline__ float redmax16(float v) {
#pragma unroll
  for (int o = 8; o >= 1; o >>= 1) v = fmaxf(v, __shfl_xor(v, o, 32));
  return v;
}
__device__ __forceinline__ float redsum16(float v) {
#pragma unroll
  for (int o = 8; o >= 1; o >>= 1) v += __shfl_xor(v, o, 32);
  return v;
}
__device__ __forceinline__ float sigmoidf_(float x) { return 1.f / (1.f + __expf(-x)); }

// ---------------------------------------------------------------- weight pre-convert
// dst[R][C] bf16 <- src[r<Rs][c<Cs] f32 (row stride ld), zero-padded outside
__global__ void k_cvt(const float* __restrict__ src, bf16_t* __restrict__ dst,
                      int R, int C, int Rs, int Cs, int ld) {
  const int i = blockIdx.x * blockDim.x + threadIdx.x;
  if (i >= R * C) return;
  const int r = i / C, c = i % C;
  const float x = (r < Rs && c < Cs) ? src[r * ld + c] : 0.f;
  dst[i] = (bf16_t)x;
}

// ---------------------------------------------------------------- kernel 1
// Fused: f=gelu(h@W1^T+b1); feat=sigmoid(f@W2^T+b2); Q,K=feat@Wq^T/Wk^T;
// charge0=sigmoid(feat@Wc^T+bc); V=h@Wv^T.  64 rows/block, 16 rows/wave.
// Hidden activations staged per-wave via TDM (tensor_load_to_lds).
__global__ __launch_bounds__(128) void k_feat(
    const float* __restrict__ hidden,
    const bf16_t* __restrict__ W1b, const float* __restrict__ b1,
    const bf16_t* __restrict__ W2b, const float* __restrict__ b2,
    const bf16_t* __restrict__ Wqb, const bf16_t* __restrict__ Wkb,
    const float* __restrict__ Wc, const float* __restrict__ bc,
    const bf16_t* __restrict__ Wvb,
    bf16_t* __restrict__ Qb, bf16_t* __restrict__ Kb, bf16_t* __restrict__ Vb,
    float* __restrict__ charges) {
  __shared__ float  hid_f32[64 * 68];    // 4 wave-private slices: 16 rows x 64 f32 (+4 pad)
  __shared__ bf16_t f_lds[64 * 72];      // 64 rows x 64 (+8 pad)
  __shared__ bf16_t feat_lds[64 * 40];   // 64 rows x 32 (cols 28..31 zero)
  const int tid = threadIdx.x;
  const int w = tid >> 5, lane = tid & 31, h = lane >> 4, nl = lane & 15;
  const int gr0 = blockIdx.x * 64;   // global row (flat B*N) of this block
  const int wr0 = w * 16;            // wave's row offset in block

  const v8f z8 = {0.f, 0.f, 0.f, 0.f, 0.f, 0.f, 0.f, 0.f};
  v8f facc[4] = {z8, z8, z8, z8};
  v8f vacc[4] = {z8, z8, z8, z8};

  for (int kc = 0; kc < 16; ++kc) {      // 16 chunks x 64 K = 1024
    const int k0 = kc * 64;
#ifdef HAVE_TDM
    {
      // TDM: 2D tile, 16 rows x 64 f32, row stride 1024, into this wave's LDS slice
      const unsigned lds_base =
          (unsigned)(uintptr_t)&hid_f32[0] + (unsigned)(wr0 * 68 * 4);
      const unsigned long long ga =
          (unsigned long long)(uintptr_t)(hidden + (size_t)(gr0 + wr0) * D_HID + k0);
      v4u g0;
      g0[0] = 1u;                                        // count=1 (valid descriptor)
      g0[1] = lds_base;                                  // lds_addr
      g0[2] = (unsigned)(ga & 0xFFFFFFFFu);              // global_addr[31:0]
      g0[3] = (unsigned)((ga >> 32) & 0x01FFFFFFu) | 0x80000000u;  // addr[56:32], type=2
      v8i g1;
      g1[0] = (int)((2u << 16) |                         // data_size = 4B
                    (1u << 20) |                         // pad_enable
                    (5u << 22) |                         // pad_interval: 64 DWORDs
                    (3u << 25));                         // pad_amount: 4 DWORDs
      g1[1] = (int)((1024u & 0xFFFFu) << 16);            // tensor_dim0 lo16
      g1[2] = (int)((1024u >> 16) | ((8192u & 0xFFFFu) << 16)); // dim0 hi | dim1 lo
      g1[3] = (int)((8192u >> 16) | (64u << 16));        // dim1 hi | tile_dim0=64
      g1[4] = 16;                                        // tile_dim1 = 16 rows
      g1[5] = 1024;                                      // tensor_dim0_stride lo32
      g1[6] = 0;
      g1[7] = 0;
      const v4i zz4 = {0, 0, 0, 0};
      const v8i zz8 = {0, 0, 0, 0, 0, 0, 0, 0};
      __builtin_amdgcn_tensor_load_to_lds(g0, g1, zz4, zz4, zz8, 0);
      __builtin_amdgcn_s_wait_tensorcnt(0);
    }
#else
    for (int t = lane; t < 16 * 16; t += 32) {           // 16 rows x 16 float4
      const int row = t >> 4, c4 = t & 15;
      reinterpret_cast<float4*>(&hid_f32[(wr0 + row) * 68])[c4] =
          reinterpret_cast<const float4*>(hidden + (size_t)(gr0 + wr0 + row) * D_HID + k0)[c4];
    }
#endif
#pragma unroll
    for (int ks = 0; ks < 2; ++ks) {
      v16bf A = load_a_f32cvt(&hid_f32[wr0 * 68 + ks * 32], 68);
      // batch all 8 weight fragments first (distinct regs -> clauses, overlap)
      v16bf Bw[4], Bv[4];
#pragma unroll
      for (int ct = 0; ct < 4; ++ct) {
        Bw[ct] = load_bT_bf16(W1b, 1024, ct * 16, k0 + ks * 32);
        Bv[ct] = load_bT_bf16(Wvb, 1024, ct * 16, k0 + ks * 32);
      }
#pragma unroll
      for (int ct = 0; ct < 4; ++ct) {
        facc[ct] = wmma_bf16(A, Bw[ct], facc[ct]);
        vacc[ct] = wmma_bf16(A, Bv[ct], vacc[ct]);
      }
    }
  }

  // gelu(exact) -> f_lds ; V -> global (bf16)
#pragma unroll
  for (int ct = 0; ct < 4; ++ct)
#pragma unroll
    for (int e = 0; e < 8; ++e) {
      const int r = e + 8 * h, col = ct * 16 + nl;
      const float x = facc[ct][e] + b1[col];
      const float gl = 0.5f * x * (1.f + erff(x * 0.70710678118f));
      f_lds[(wr0 + r) * 72 + col] = (bf16_t)gl;
      Vb[(gr0 + wr0 + r) * 64 + col] = (bf16_t)vacc[ct][e];
    }

  // features = sigmoid(f @ W2^T + b2)   [rows,28] padded to 32 (W2b pre-padded)
  v8f a2[2] = {z8, z8};
#pragma unroll
  for (int ks = 0; ks < 2; ++ks) {
    v16bf A  = load_a_bf16(&f_lds[wr0 * 72 + ks * 32], 72);
    v16bf B0 = load_bT_bf16(W2b, 64, 0,  ks * 32);
    v16bf B1 = load_bT_bf16(W2b, 64, 16, ks * 32);
    a2[0] = wmma_bf16(A, B0, a2[0]);
    a2[1] = wmma_bf16(A, B1, a2[1]);
  }
#pragma unroll
  for (int ct = 0; ct < 2; ++ct)
#pragma unroll
    for (int e = 0; e < 8; ++e) {
      const int r = e + 8 * h, col = ct * 16 + nl;
      const float s = (col < 28) ? sigmoidf_(a2[ct][e] + b2[col]) : 0.f;
      feat_lds[(wr0 + r) * 40 + col] = (bf16_t)s;
    }

  // Q, K  (K-dim = 28 zero-padded to 32 in Wqb/Wkb; single wmma k-step)
  {
    v16bf Af = load_a_bf16(&feat_lds[wr0 * 40], 40);
    v16bf Bq[4], Bk[4];
#pragma unroll
    for (int ct = 0; ct < 4; ++ct) {
      Bq[ct] = load_bT_bf16(Wqb, 32, ct * 16, 0);
      Bk[ct] = load_bT_bf16(Wkb, 32, ct * 16, 0);
    }
#pragma unroll
    for (int ct = 0; ct < 4; ++ct) {
      v8f q  = wmma_bf16(Af, Bq[ct], z8);
      v8f kk = wmma_bf16(Af, Bk[ct], z8);
#pragma unroll
      for (int e = 0; e < 8; ++e) {
        const int r = e + 8 * h, col = ct * 16 + nl;
        Qb[(gr0 + wr0 + r) * 64 + col] = (bf16_t)q[e];
        Kb[(gr0 + wr0 + r) * 64 + col] = (bf16_t)kk[e];
      }
    }
  }

  __syncthreads();
  if (tid < 64) {   // charge0 per row
    float acc = bc[0];
    for (int k = 0; k < 28; ++k) acc += (float)feat_lds[tid * 40 + k] * Wc[k];
    charges[gr0 + tid] = sigmoidf_(acc);
  }
}

// logits_t[n,m] = compat[n,m]*(1 + step*sum_{s=1..npast} c_s[n]c_s[m]), causal else NEG
__device__ __forceinline__ void softmax_stats(
    v16bf qa0, v16bf qa1, const bf16_t* __restrict__ Kb,
    const float* __restrict__ charges, int bn, int r0, int npast, float step,
    const float crow[4][8], float rmax[8], float rsum[8]) {
  const int lane = threadIdx.x & 31, h = lane >> 4, nl = lane & 15;
#pragma unroll
  for (int e = 0; e < 8; ++e) { rmax[e] = -3.0e38f; rsum[e] = 0.f; }
  const int mtiles = r0 / 16 + 1;
  for (int mt = 0; mt < mtiles; ++mt) {
    const int m0 = mt * 16;
    __builtin_prefetch(Kb + (size_t)(bn + m0 + 16) * 64, 0, 1);
    v8f qk = qk_tile_pre(qa0, qa1, Kb, bn + m0);
    float ccol[4];
    for (int s = 0; s < npast; ++s) ccol[s] = charges[(s + 1) * BN + bn + m0 + nl];
#pragma unroll
    for (int e = 0; e < 8; ++e) {
      const int n = r0 + e + 8 * h, m = m0 + nl;
      float dot = 0.f;
      for (int s = 0; s < npast; ++s) dot += crow[s][e] * ccol[s];
      float v = qk[e] * 0.125f * (1.f + step * dot);
      v = (m <= n) ? v : NEGV;
      const float tmax = redmax16(v);
      const float mnew = fmaxf(rmax[e], tmax);
      const float p = __expf(v - mnew);
      const float ts = redsum16(p);
      rsum[e] = rsum[e] * __expf(rmax[e] - mnew) + ts;
      rmax[e] = mnew;
    }
  }
}

// ---------------------------------------------------------------- kernel 2
// one refinement pass: softmax row stats + column sums -> received
__global__ __launch_bounds__(128) void k_pass(
    const bf16_t* __restrict__ Qb, const bf16_t* __restrict__ Kb,
    const float* __restrict__ charges, float* __restrict__ received,
    const float* __restrict__ step_p, int npast) {
  const int b = blockIdx.x;
  const int w = threadIdx.x >> 5, lane = threadIdx.x & 31, h = lane >> 4, nl = lane & 15;
  const int r0 = blockIdx.y * 64 + w * 16;
  const int bn = b * N_TOK;
  const float step = *step_p;
  float crow[4][8];
  for (int s = 0; s < npast; ++s)
#pragma unroll
    for (int e = 0; e < 8; ++e)
      crow[s][e] = charges[(s + 1) * BN + bn + r0 + e + 8 * h];

  // Q fragments are invariant across the whole m stream: load once
  const v16bf qa0 = load_a_bf16(Qb + (size_t)(bn + r0) * 64, 64);
  const v16bf qa1 = load_a_bf16(Qb + (size_t)(bn + r0) * 64 + 32, 64);

  float rmax[8], rsum[8];
  softmax_stats(qa0, qa1, Kb, charges, bn, r0, npast, step, crow, rmax, rsum);

  // second sweep: column partial sums of attn, atomically accumulated
  const int mtiles = r0 / 16 + 1;
  for (int mt = 0; mt < mtiles; ++mt) {
    const int m0 = mt * 16;
    __builtin_prefetch(Kb + (size_t)(bn + m0 + 16) * 64, 0, 1);
    v8f qk = qk_tile_pre(qa0, qa1, Kb, bn + m0);
    float ccol[4];
    for (int s = 0; s < npast; ++s) ccol[s] = charges[(s + 1) * BN + bn + m0 + nl];
    float colp = 0.f;
#pragma unroll
    for (int e = 0; e < 8; ++e) {
      const int n = r0 + e + 8 * h, m = m0 + nl;
      float dot = 0.f;
      for (int s = 0; s < npast; ++s) dot += crow[s][e] * ccol[s];
      float v = qk[e] * 0.125f * (1.f + step * dot);
      v = (m <= n) ? v : NEGV;
      colp += __expf(v - rmax[e]) / rsum[e];
    }
    colp += __shfl_xor(colp, 16, 32);
    if (h == 0) atomicAdd(&received[bn + m0 + nl], colp);
  }
}

// ---------------------------------------------------------------- small kernels
__global__ void k_zero(float* __restrict__ p) {
  const int g = blockIdx.x * blockDim.x + threadIdx.x;
  if (g < BN) p[g] = 0.f;
}
__global__ void k_charge(const float* __restrict__ received, float* __restrict__ charges,
                         const float* __restrict__ decay_p, int it) {
  const int g = blockIdx.x * blockDim.x + threadIdx.x;
  if (g >= BN) return;
  const float decay = *decay_p;
  const float s = sigmoidf_(received[g] - 1.f);
  charges[(it + 1) * BN + g] = charges[it * BN + g] * (1.f - decay * s);
}

// ---------------------------------------------------------------- kernel 3
// final: out = (softmax(logits_4) @ V) @ Wo^T * 0.1
__global__ __launch_bounds__(128) void k_final(
    const bf16_t* __restrict__ Qb, const bf16_t* __restrict__ Kb,
    const bf16_t* __restrict__ Vb, const bf16_t* __restrict__ Wob,
    const float* __restrict__ charges, const float* __restrict__ step_p,
    float* __restrict__ out) {
  __shared__ bf16_t p_lds[4][16 * 40];   // per-wave P tile 16x32 (padded)
  __shared__ bf16_t o_lds[4][16 * 72];   // per-wave O tile 16x64 (padded)
  const int b = blockIdx.x;
  const int w = threadIdx.x >> 5, lane = threadIdx.x & 31, h = lane >> 4, nl = lane & 15;
  const int r0 = blockIdx.y * 64 + w * 16;
  const int bn = b * N_TOK;
  const float step = *step_p;
  const int npast = 4;

  float crow[4][8];
  for (int s = 0; s < npast; ++s)
#pragma unroll
    for (int e = 0; e < 8; ++e)
      crow[s][e] = charges[(s + 1) * BN + bn + r0 + e + 8 * h];

  const v16bf qa0 = load_a_bf16(Qb + (size_t)(bn + r0) * 64, 64);
  const v16bf qa1 = load_a_bf16(Qb + (size_t)(bn + r0) * 64 + 32, 64);

  float rmax[8], rsum[8];
  softmax_stats(qa0, qa1, Kb, charges, bn, r0, npast, step, crow, rmax, rsum);

  const v8f z8 = {0.f, 0.f, 0.f, 0.f, 0.f, 0.f, 0.f, 0.f};
  v8f oacc[4] = {z8, z8, z8, z8};
  const int m32 = r0 / 32 + 1;
  for (int mt = 0; mt < m32; ++mt) {
    const int m0 = mt * 32;
    __builtin_prefetch(Vb + (size_t)(bn + m0 + 32) * 64, 0, 1);
#pragma unroll
    for (int sub = 0; sub < 2; ++sub) {
      const int ms = m0 + sub * 16;
      v8f qk = qk_tile_pre(qa0, qa1, Kb, bn + ms);
      float ccol[4];
      for (int s = 0; s < npast; ++s) ccol[s] = charges[(s + 1) * BN + bn + ms + nl];
#pragma unroll
      for (int e = 0; e < 8; ++e) {
        const int n = r0 + e + 8 * h, m = ms + nl;
        float dot = 0.f;
        for (int s = 0; s < npast; ++s) dot += crow[s][e] * ccol[s];
        float v = qk[e] * 0.125f * (1.f + step * dot);
        v = (m <= n) ? v : NEGV;
        const float p = __expf(v - rmax[e]);   // masked -> 0
        p_lds[w][(e + 8 * h) * 40 + sub * 16 + nl] = (bf16_t)p;
      }
    }
    v16bf Ap = load_a_bf16(&p_lds[w][0], 40);  // 16 rows x 32 m (=K)
    v16bf Bv[4];
#pragma unroll
    for (int ct = 0; ct < 4; ++ct) Bv[ct] = load_b_bf16(Vb, 64, bn + m0, ct * 16);
#pragma unroll
    for (int ct = 0; ct < 4; ++ct) oacc[ct] = wmma_bf16(Ap, Bv[ct], oacc[ct]);
  }

  // normalize and re-layout O (C-layout -> A-layout via LDS)
#pragma unroll
  for (int ct = 0; ct < 4; ++ct)
#pragma unroll
    for (int e = 0; e < 8; ++e)
      o_lds[w][(e + 8 * h) * 72 + ct * 16 + nl] = (bf16_t)(oacc[ct][e] / rsum[e]);

  // out = O @ Wo^T * 0.1   (16 x 1024); O fragments invariant across dt tiles
  const v16bf oa0 = load_a_bf16(&o_lds[w][0],  72);
  const v16bf oa1 = load_a_bf16(&o_lds[w][32], 72);
  for (int dt = 0; dt < 64; ++dt) {
    v16bf B0 = load_bT_bf16(Wob, 64, dt * 16, 0);
    v16bf B1 = load_bT_bf16(Wob, 64, dt * 16, 32);
    v8f acc = z8;
    acc = wmma_bf16(oa0, B0, acc);
    acc = wmma_bf16(oa1, B1, acc);
#pragma unroll
    for (int e = 0; e < 8; ++e) {
      const int r = e + 8 * h;
      out[(size_t)(bn + r0 + r) * D_HID + dt * 16 + nl] = 0.1f * acc[e];
    }
  }
}

// ---------------------------------------------------------------- host launcher
extern "C" void kernel_launch(void* const* d_in, const int* in_sizes, int n_in,
                              void* d_out, int out_size, void* d_ws, size_t ws_size,
                              hipStream_t stream) {
  (void)in_sizes; (void)n_in; (void)out_size; (void)ws_size;
  const float* hidden = (const float*)d_in[0];
  const float* W1 = (const float*)d_in[1];
  const float* b1 = (const float*)d_in[2];
  const float* W2 = (const float*)d_in[3];
  const float* b2 = (const float*)d_in[4];
  const float* Wq = (const float*)d_in[5];
  const float* Wk = (const float*)d_in[6];
  const float* Wc = (const float*)d_in[7];
  const float* bc = (const float*)d_in[8];
  const float* Wv = (const float*)d_in[9];
  const float* Wo = (const float*)d_in[10];
  const float* step_p  = (const float*)d_in[11];
  const float* decay_p = (const float*)d_in[12];

  char* ws = (char*)d_ws;
  bf16_t* Qb = (bf16_t*)ws; ws += (size_t)BN * 64 * sizeof(bf16_t);
  bf16_t* Kb = (bf16_t*)ws; ws += (size_t)BN * 64 * sizeof(bf16_t);
  bf16_t* Vb = (bf16_t*)ws; ws += (size_t)BN * 64 * sizeof(bf16_t);
  float* charges  = (float*)ws; ws += (size_t)5 * BN * sizeof(float);  // c0..c4
  float* received = (float*)ws; ws += (size_t)BN * sizeof(float);
  bf16_t* W1b = (bf16_t*)ws; ws += (size_t)64 * 1024 * sizeof(bf16_t);
  bf16_t* Wvb = (bf16_t*)ws; ws += (size_t)64 * 1024 * sizeof(bf16_t);
  bf16_t* Wob = (bf16_t*)ws; ws += (size_t)1024 * 64 * sizeof(bf16_t);
  bf16_t* W2b = (bf16_t*)ws; ws += (size_t)32 * 64 * sizeof(bf16_t);
  bf16_t* Wqb = (bf16_t*)ws; ws += (size_t)64 * 32 * sizeof(bf16_t);
  bf16_t* Wkb = (bf16_t*)ws; ws += (size_t)64 * 32 * sizeof(bf16_t);
  float* outp = (float*)d_out;

  // one-time weight conversion to bf16 (pre-padded where needed)
  k_cvt<<<(64 * 1024 + 255) / 256, 256, 0, stream>>>(W1, W1b, 64, 1024, 64, 1024, 1024);
  k_cvt<<<(64 * 1024 + 255) / 256, 256, 0, stream>>>(Wv, Wvb, 64, 1024, 64, 1024, 1024);
  k_cvt<<<(1024 * 64 + 255) / 256, 256, 0, stream>>>(Wo, Wob, 1024, 64, 1024, 64, 64);
  k_cvt<<<(32 * 64 + 255) / 256, 256, 0, stream>>>(W2, W2b, 32, 64, 28, 64, 64);
  k_cvt<<<(64 * 32 + 255) / 256, 256, 0, stream>>>(Wq, Wqb, 64, 32, 64, 28, 28);
  k_cvt<<<(64 * 32 + 255) / 256, 256, 0, stream>>>(Wk, Wkb, 64, 32, 64, 28, 28);

  k_feat<<<BN / 64, 128, 0, stream>>>(hidden, W1b, b1, W2b, b2, Wqb, Wkb, Wc, bc, Wvb,
                                      Qb, Kb, Vb, charges);
  for (int i = 0; i < 4; ++i) {
    k_zero<<<BN / 256, 256, 0, stream>>>(received);
    k_pass<<<dim3(4, N_TOK / 64), 128, 0, stream>>>(Qb, Kb, charges, received, step_p, i);
    k_charge<<<BN / 256, 256, 0, stream>>>(received, charges, decay_p, i);
  }
  k_final<<<dim3(4, N_TOK / 64), 128, 0, stream>>>(Qb, Kb, Vb, Wob, charges, step_p, outp);
}
